// SelfAttention_68856915689768
// MI455X (gfx1250) — matrix operand order repaired and checked by reading, not compiled, not run on hardware
//
#include <hip/hip_runtime.h>
#include <hip/hip_bf16.h>

// ---------------------------------------------------------------------------
// MI455X (gfx1250) self-attention: bf16 WMMA GEMMs (TDM-staged weights in LDS)
// + flash attention. Matrix math via v_wmma_f32_16x16x32_bf16 (wave32).
// ---------------------------------------------------------------------------

typedef __attribute__((ext_vector_type(16))) __bf16 v16bf;
typedef __attribute__((ext_vector_type(8)))  __bf16 v8bf;
typedef __attribute__((ext_vector_type(8)))  float  v8f;

// TDM descriptor groups (amdgpu-toolchain / clang-23 6-arg builtin form)
typedef __attribute__((ext_vector_type(4))) unsigned int tdm_g0_t;
typedef __attribute__((ext_vector_type(8))) int          tdm_g1_t;
typedef __attribute__((ext_vector_type(4))) int          tdm_g2_t;
typedef __attribute__((ext_vector_type(8))) int          tdm_g4_t;

__device__ __forceinline__ v8f wmma_bf16(v16bf a, v16bf b, v8f c) {
  // (neg_a, A, neg_b, B, c_mod, C, reuse_a, reuse_b)
  return __builtin_amdgcn_wmma_f32_16x16x32_bf16(false, a, false, b,
                                                 (short)0, c, false, false);
}

// A-fragment loader (16-bit A 16x32 layout, ISA 7.12.2):
// per lane: elements 0..7 <- p[0..7], elements 8..15 <- p[16..23].
__device__ __forceinline__ v16bf load_split16(const __bf16* p) {
  const v8bf lo = *reinterpret_cast<const v8bf*>(p);
  const v8bf hi = *reinterpret_cast<const v8bf*>(p + 16);
  v16bf r;
#pragma unroll
  for (int i = 0; i < 8; ++i) { r[i] = lo[i]; r[i + 8] = hi[i]; }
  return r;
}

// 16 contiguous bf16 as two 16B loads (16B alignment guaranteed by caller).
__device__ __forceinline__ v16bf load_contig16(const void* p) {
  const v8bf lo = *reinterpret_cast<const v8bf*>(p);
  const v8bf hi = *reinterpret_cast<const v8bf*>((const char*)p + 16);
  v16bf r;
#pragma unroll
  for (int i = 0; i < 8; ++i) { r[i] = lo[i]; r[i + 8] = hi[i]; }
  return r;
}

// ---------------------------------------------------------------------------
// TDM: load a 2D bf16 tile (tile_w x tile_h, row stride in elements) from
// global memory into LDS, with 16B padding after every 1024 data bytes
// (pad_interval=7 -> 256 DWORDs, pad_amount=3 -> 4 DWORDs) to de-align LDS
// bank strides. Effective LDS row stride = row_bytes + row_bytes/1024*16.
// ---------------------------------------------------------------------------
__device__ __forceinline__ void tdm_load_2d_to_lds(unsigned lds_addr,
                                                   const void* gptr,
                                                   unsigned tile_w_elems,
                                                   unsigned tile_h_rows,
                                                   unsigned row_stride_elems) {
  const unsigned long long ga = (unsigned long long)(uintptr_t)gptr;
  tdm_g0_t g0;
  g0[0] = 1u;                                   // count=1 (valid user D#)
  g0[1] = lds_addr;                             // lds_addr  [63:32]
  g0[2] = (unsigned)ga;                         // global_addr low 32
  g0[3] = (unsigned)((ga >> 32) & 0x01FFFFFFu)  // global_addr [56:32]
          | (2u << 30);                         // type = 2 ("image")
  tdm_g1_t g1;
  // data_size=1 (2B), pad_enable=1, pad_interval=7 (256 dw), pad_amount=3 (4 dw)
  g1[0] = (int)((1u << 16) | (1u << 20) | (7u << 22) | (3u << 25));
  const unsigned td0 = row_stride_elems;        // tensor_dim0
  const unsigned td1 = tile_h_rows;             // tensor_dim1 (tile in-bounds)
  g1[1] = (int)((td0 & 0xFFFFu) << 16);                  // dim0[15:0] @ 63:48
  g1[2] = (int)((td0 >> 16) | ((td1 & 0xFFFFu) << 16));  // dim0 hi | dim1 lo
  g1[3] = (int)((td1 >> 16) | (tile_w_elems << 16));     // dim1 hi | tile_dim0
  g1[4] = (int)(tile_h_rows & 0xFFFFu);                  // tile_dim1 (tile_dim2=0)
  g1[5] = (int)row_stride_elems;                         // dim0_stride low 32
  g1[6] = 0;                                             // dim0_stride hi | dim1_stride lo
  g1[7] = 0;
  const tdm_g2_t gz4 = {0, 0, 0, 0};
  const tdm_g4_t gz8 = {0, 0, 0, 0, 0, 0, 0, 0};
  __builtin_amdgcn_tensor_load_to_lds(g0, g1, gz4, gz4, gz8, 0);
}

// ---------------------------------------------------------------------------
// Prep kernels
// ---------------------------------------------------------------------------
__global__ void f32_to_bf16_kernel(const float* __restrict__ in,
                                   __bf16* __restrict__ out, int n) {
  int i = blockIdx.x * blockDim.x + threadIdx.x;
  if (i < n) out[i] = (__bf16)in[i];
}

// in: R x C (row major) fp32  ->  out: C x R (row major) bf16
__global__ void transpose_to_bf16_kernel(const float* __restrict__ in,
                                         __bf16* __restrict__ out,
                                         int R, int C) {
  int idx = blockIdx.x * blockDim.x + threadIdx.x;   // index over output
  if (idx >= R * C) return;
  int c = idx / R;
  int r = idx - c * R;
  out[idx] = (__bf16)in[(size_t)r * C + c];
}

// ---------------------------------------------------------------------------
// WMMA GEMM:  D[M,N] = A[M,K](bf16) * Bt[N,K]^T(bf16) + bias[N],  K = 1024.
// The block's 64-row weight slab Bt[n0..n0+63, 0..K) is TDM-staged into LDS
// once, then all 8 waves stream A from global and read B fragments from LDS.
// mode 0: store fp32 to Cout ; mode 1: scatter bf16 into Q / K / V^T.
// Block: 256 threads = 8 waves, 4(M) x 2(N), 32x32 per wave -> 128x64 tile.
// ---------------------------------------------------------------------------
#define GEMM_K      1024
#define B_ROW_BYTES (GEMM_K * 2)                       // 2048 data bytes
#define B_ROW_LDS   (B_ROW_BYTES + (B_ROW_BYTES / 1024) * 16)  // 2080 padded
#define B_LDS_BYTES (64 * B_ROW_LDS)                   // 133120

__global__ __launch_bounds__(256)
void wmma_gemm_kernel(const __bf16* __restrict__ A, const __bf16* __restrict__ Bt,
                      const float* __restrict__ bias, int M, int N,
                      int mode, float* __restrict__ Cout,
                      __bf16* __restrict__ Qb, __bf16* __restrict__ Kb,
                      __bf16* __restrict__ Vt) {
  constexpr int K = GEMM_K;
  extern __shared__ char bsm[];                 // padded 64 x K bf16 slab

  const int wave = threadIdx.x >> 5;
  const int lane = threadIdx.x & 31;
  const int l15  = lane & 15;
  const int lh   = lane >> 4;

  const int m0    = blockIdx.x * 128 + (wave & 3) * 32;
  const int n0blk = blockIdx.y * 64;
  const int nw    = (wave >> 2) * 32;           // local N offset within slab

  // ---- stage weight slab via Tensor Data Mover (wave 0 issues; EXEC ignored)
  if (threadIdx.x < 32) {
    tdm_load_2d_to_lds((unsigned)(uintptr_t)bsm,
                       Bt + (size_t)n0blk * K,
                       /*tile_w=*/K, /*tile_h=*/64, /*row_stride=*/K);
    __builtin_amdgcn_s_wait_tensorcnt((short)0);
  }
  __syncthreads();

  const __bf16* arow0 = A + (size_t)(m0 + l15)      * K;
  const __bf16* arow1 = A + (size_t)(m0 + 16 + l15) * K;
  const char*   brow0 = bsm + (size_t)(nw + l15)      * B_ROW_LDS;
  const char*   brow1 = bsm + (size_t)(nw + 16 + l15) * B_ROW_LDS;

  v8f acc[2][2] = {};
  for (int k = 0; k < K; k += 32) {
    const int ka    = k + 16 * lh;                    // A half-wave K offset
    const int kb    = k + 16 * lh;                    // B half-wave K offset
    const int kboff = kb * 2 + ((kb >> 9) << 4);      // pad-aware LDS offset
    v16bf a0 = load_contig16(arow0 + ka);
    v16bf a1 = load_contig16(arow1 + ka);
    v16bf b0 = load_contig16(brow0 + kboff);
    v16bf b1 = load_contig16(brow1 + kboff);
    acc[0][0] = wmma_bf16(a0, b0, acc[0][0]);
    acc[0][1] = wmma_bf16(a0, b1, acc[0][1]);
    acc[1][0] = wmma_bf16(a1, b0, acc[1][0]);
    acc[1][1] = wmma_bf16(a1, b1, acc[1][1]);
  }

  const int n0 = n0blk + nw;
#pragma unroll
  for (int mi = 0; mi < 2; ++mi)
#pragma unroll
    for (int ni = 0; ni < 2; ++ni)
#pragma unroll
      for (int e = 0; e < 8; ++e) {
        const int row = m0 + 16 * mi + e + 8 * lh;    // C layout: M = r + 8*half
        const int col = n0 + 16 * ni + l15;           // N = lane&15
        const float v = acc[mi][ni][e] + bias[col];
        if (mode == 0) {
          Cout[(size_t)row * N + col] = v;
        } else {
          // row -> (b, s); col -> (section, head, d) ; D=1024, S=2048, H=16
          const int b   = row >> 11;
          const int s   = row & 2047;
          const int sec = col >> 10;
          const int dD  = col & 1023;
          const int h   = dD >> 6;
          const int d   = dD & 63;
          const int bh  = b * 16 + h;
          const __bf16 bv = (__bf16)v;
          if (sec == 0)      Qb[((size_t)bh * 2048 + s) * 64 + d] = bv;
          else if (sec == 1) Kb[((size_t)bh * 2048 + s) * 64 + d] = bv;
          else               Vt[((size_t)bh * 64 + d) * 2048 + s] = bv;
        }
      }
}

// ---------------------------------------------------------------------------
// Flash attention (causal), bf16 WMMA, fp32 online softmax.
// Grid: (S/64, B*H), block 128 threads = 4 waves; each wave owns 16 q-rows.
// ---------------------------------------------------------------------------
__global__ __launch_bounds__(128)
void flash_attn_kernel(const __bf16* __restrict__ Qb, const __bf16* __restrict__ Kb,
                       const __bf16* __restrict__ Vt, __bf16* __restrict__ Ob,
                       const int* __restrict__ causal_flag) {
  constexpr int S = 2048, DH = 64, H = 16;
  const int wave = threadIdx.x >> 5;
  const int lane = threadIdx.x & 31;
  const int l15  = lane & 15;
  const int lh   = lane >> 4;
  const int bh   = blockIdx.y;           // b*H + h
  const int b    = bh >> 4;
  const int h    = bh & 15;
  const int q0   = blockIdx.x * 64 + wave * 16;
  const bool causal = (*causal_flag) != 0;

  __shared__ __bf16 pshared[4][16 * 32];
  __bf16* plds = pshared[wave];

  const __bf16* qbase = Qb + ((size_t)bh * S + (q0 + l15)) * DH;
  v16bf qf[2];
#pragma unroll
  for (int t = 0; t < 2; ++t) qf[t] = load_contig16(qbase + 32 * t + 16 * lh);

  v8f o[4] = {};
  float m_r[8], l_r[8];
#pragma unroll
  for (int e = 0; e < 8; ++e) { m_r[e] = -__builtin_inff(); l_r[e] = 0.f; }

  const __bf16* kbh = Kb + (size_t)bh * S * DH;
  const __bf16* vbh = Vt + (size_t)bh * DH * S;

  const int kend = causal ? (q0 + 16) : S;
  for (int kb = 0; kb < kend; kb += 32) {
    // ---- scores: two 16x16 tiles (cols kb..+15, kb+16..+31) --------------
    v8f s0 = {}, s1 = {};
#pragma unroll
    for (int t = 0; t < 2; ++t) {
      v16bf kf0 = load_contig16(kbh + (size_t)(kb + l15) * DH + 32 * t + 16 * lh);
      v16bf kf1 = load_contig16(kbh + (size_t)(kb + 16 + l15) * DH + 32 * t + 16 * lh);
      s0 = wmma_bf16(qf[t], kf0, s0);
      s1 = wmma_bf16(qf[t], kf1, s1);
    }

    // ---- scale, mask, online softmax -------------------------------------
    float alpha[8];
#pragma unroll
    for (int e = 0; e < 8; ++e) {
      const int row = q0 + e + 8 * lh;          // C layout row
      float x0 = s0[e] * 0.125f;                // 1/sqrt(64)
      float x1 = s1[e] * 0.125f;
      if (causal) {
        if (kb + l15 > row)      x0 = -__builtin_inff();
        if (kb + 16 + l15 > row) x1 = -__builtin_inff();
      }
      float mx = fmaxf(x0, x1);
#pragma unroll
      for (int off = 1; off < 16; off <<= 1)
        mx = fmaxf(mx, __shfl_xor(mx, off, 32));
      const float mnew = fmaxf(m_r[e], mx);
      const float a  = __expf(m_r[e] - mnew);
      const float p0 = __expf(x0 - mnew);
      const float p1 = __expf(x1 - mnew);
      float rs = p0 + p1;
#pragma unroll
      for (int off = 1; off < 16; off <<= 1)
        rs += __shfl_xor(rs, off, 32);
      l_r[e] = l_r[e] * a + rs;
      m_r[e] = mnew;
      alpha[e] = a;
      const int prow = e + 8 * lh;              // C layout -> LDS row-major P
      plds[prow * 32 + l15]      = (__bf16)p0;
      plds[prow * 32 + 16 + l15] = (__bf16)p1;
    }

#pragma unroll
    for (int j = 0; j < 4; ++j)
#pragma unroll
      for (int e = 0; e < 8; ++e) o[j][e] *= alpha[e];

    // wave-internal LDS pipe is in-order; stop compiler reordering only.
    __builtin_amdgcn_wave_barrier();

    // ---- P as A-fragment, PV accumulate ----------------------------------
    v16bf pf = load_contig16(plds + (size_t)l15 * 32 + 16 * lh);
#pragma unroll
    for (int j = 0; j < 4; ++j) {
      const v16bf vf = load_contig16(vbh + (size_t)(16 * j + l15) * S + kb + 16 * lh);
      o[j] = wmma_bf16(pf, vf, o[j]);
    }
  }

  // ---- normalize + store bf16 attention output [B,S,D] -------------------
#pragma unroll
  for (int e = 0; e < 8; ++e) {
    const float inv = 1.0f / l_r[e];
    const int row = q0 + e + 8 * lh;
    __bf16* orow = Ob + ((size_t)b * S + row) * (H * DH) + h * DH;
#pragma unroll
    for (int j = 0; j < 4; ++j)
      orow[16 * j + l15] = (__bf16)(o[j][e] * inv);
  }
}

// ---------------------------------------------------------------------------
// Host-side launch
// ---------------------------------------------------------------------------
extern "C" void kernel_launch(void* const* d_in, const int* in_sizes, int n_in,
                              void* d_out, int out_size, void* d_ws, size_t ws_size,
                              hipStream_t stream) {
  const float* x     = (const float*)d_in[0];
  const float* W_in  = (const float*)d_in[1];
  const float* b_in  = (const float*)d_in[2];
  const float* W_out = (const float*)d_in[3];
  const float* b_out = (const float*)d_in[4];
  const int*   cmask = (const int*)d_in[5];
  (void)in_sizes; (void)n_in; (void)out_size; (void)ws_size;

  constexpr int B = 4, S = 2048, D = 1024, H = 16, DH = 64;
  constexpr int M  = B * S;      // 8192
  constexpr int N1 = 3 * D;      // 3072

  char* ws = (char*)d_ws;
  size_t off = 0;
  auto take = [&](size_t bytes) -> char* {
    char* p = ws + off;
    off = (off + bytes + 255) & ~(size_t)255;
    return p;
  };
  __bf16* xb    = (__bf16*)take((size_t)M * D * 2);            // 16 MiB
  __bf16* WtIn  = (__bf16*)take((size_t)N1 * D * 2);           //  6 MiB
  __bf16* WtOut = (__bf16*)take((size_t)D * D * 2);            //  2 MiB
  __bf16* Qb    = (__bf16*)take((size_t)B * H * S * DH * 2);   // 16 MiB
  __bf16* Kb    = (__bf16*)take((size_t)B * H * S * DH * 2);   // 16 MiB
  __bf16* Vt    = (__bf16*)take((size_t)B * H * DH * S * 2);   // 16 MiB
  __bf16* Ab    = (__bf16*)take((size_t)M * D * 2);            // 16 MiB

  // 1) precision / layout prep
  {
    int n = M * D;
    f32_to_bf16_kernel<<<dim3((n + 255) / 256), dim3(256), 0, stream>>>(x, xb, n);
  }
  {
    int n = D * N1;
    transpose_to_bf16_kernel<<<dim3((n + 255) / 256), dim3(256), 0, stream>>>(
        W_in, WtIn, D, N1);
  }
  {
    int n = D * D;
    transpose_to_bf16_kernel<<<dim3((n + 255) / 256), dim3(256), 0, stream>>>(
        W_out, WtOut, D, D);
  }

  // 2) QKV projection (TDM-staged weights), scatter into Q / K / V^T layouts
  wmma_gemm_kernel<<<dim3(M / 128, N1 / 64), dim3(256), B_LDS_BYTES, stream>>>(
      xb, WtIn, b_in, M, N1, /*mode=*/1, nullptr, Qb, Kb, Vt);

  // 3) causal flash attention
  flash_attn_kernel<<<dim3(S / 64, B * H), dim3(128), 0, stream>>>(
      Qb, Kb, Vt, Ab, cmask);

  // 4) output projection (TDM-staged weights) -> fp32 d_out
  wmma_gemm_kernel<<<dim3(M / 128, D / 64), dim3(256), B_LDS_BYTES, stream>>>(
      Ab, WtOut, b_out, M, D, /*mode=*/0, (float*)d_out,
      nullptr, nullptr, nullptr);
}
